// MultiHeadedAttention_8985071583467
// MI455X (gfx1250) — matrix operand order repaired and checked by reading, not compiled, not run on hardware
//
#include <hip/hip_runtime.h>

// ---------------------------------------------------------------------------
// MultiHeadedAttention forward for MI455X (gfx1250, wave32, WMMA).
// out  = (ctx @ wo^T + bo)             [B,S,1024] fp32
// attn = softmax(mask(QK^T/8))         [B,H,S,S]  fp32  (second tuple output)
// All matmuls via v_wmma_f32_16x16x32_bf16 (bf16 inputs, fp32 accum).
// fp32 inputs/weights are pre-converted to bf16 once (they are re-read 16-32x
// by the GEMM tiling, so this halves GEMM read bandwidth and removes all
// v_cvt work from the WMMA inner loops).
// ---------------------------------------------------------------------------

typedef __attribute__((ext_vector_type(16))) __bf16 v16bf;
typedef __attribute__((ext_vector_type(8)))  float  v8f;

constexpr int Hn = 16;    // heads
constexpr int Dm = 1024;  // d_model
constexpr int Dk = 64;    // d_k
constexpr int Bn = 2;     // batch
constexpr int Sn = 2048;  // sequence

// ---- WMMA fragment loaders (contiguous K order, same for A and B) ------------------
// A 16x32 bf16: lane holds M=lane&15; elems 0..15 -> K=hiA+e
// where hiA = (lane>=16)*16.  p points at &src[row*ld + k0 + hiA].
__device__ __forceinline__ v16bf load_a_bf16(const __bf16* __restrict__ p) {
  v16bf a;
#pragma unroll
  for (int e = 0; e < 16; ++e) a[e] = p[e];
  return a;
}
// B 32x16 bf16: lane holds N=lane&15; elems 0..15 -> K=hiB+e, hiB=(lane>=16)*16.
// p points at &src_column_n[k0 + hiB] (contiguous for X@W^T-style operands).
__device__ __forceinline__ v16bf load_b_bf16(const __bf16* __restrict__ p) {
  v16bf b;
#pragma unroll
  for (int e = 0; e < 16; ++e) b[e] = p[e];
  return b;
}

// ---------------------------------------------------------------------------
// fp32 -> bf16 elementwise conversion (vectorized by 4).
// ---------------------------------------------------------------------------
__global__ __launch_bounds__(256) void f32_to_bf16_kernel(
    const float* __restrict__ src, __bf16* __restrict__ dst, long n) {
  const long i = ((long)blockIdx.x * 256 + threadIdx.x) * 4;
  if (i + 3 < n) {
    const float4 f = *(const float4*)(src + i);
    dst[i + 0] = (__bf16)f.x;
    dst[i + 1] = (__bf16)f.y;
    dst[i + 2] = (__bf16)f.z;
    dst[i + 3] = (__bf16)f.w;
  } else {
    for (long j = i; j < n; ++j) dst[j] = (__bf16)src[j];
  }
}

// ---------------------------------------------------------------------------
// GEMM: Y[M,N] = X[M,K] @ W[N,K]^T + bias[N], X and W bf16, fp32 accumulate.
// Workgroup = 8 waves = 128 rows x 64 cols tile; each wave: 16x64 strip.
// HEAD_SPLIT: write bf16 into [B,H,S,Dk] layout (for Q/K/V); else fp32 [M,N].
// ---------------------------------------------------------------------------
template <bool HEAD_SPLIT>
__global__ __launch_bounds__(256) void gemm_bf16_kernel(
    const __bf16* __restrict__ X, const __bf16* __restrict__ W,
    const float* __restrict__ bias, float* __restrict__ Yf,
    __bf16* __restrict__ Yb, int M, int N, int K) {
  const int lane  = threadIdx.x & 31;
  const int wave  = threadIdx.x >> 5;
  const int tileN = blockIdx.x * 64;
  const int tileM = blockIdx.y * 128 + wave * 16;
  const int lidx  = lane & 15;
  const int hiA   = (lane >> 4) * 16;
  const int hiB   = (lane >> 4) * 16;

  const __bf16* __restrict__ xrow = X + (long)(tileM + lidx) * K + hiA;

  v8f c[4] = {v8f{}, v8f{}, v8f{}, v8f{}};

  for (int k0 = 0; k0 < K; k0 += 32) {
    // prefetch the A operand one K-tile ahead (-> global_prefetch_b8)
    __builtin_prefetch(xrow + k0 + 32, 0, 1);
    const v16bf a = load_a_bf16(xrow + k0);
#pragma unroll
    for (int t = 0; t < 4; ++t) {
      const v16bf bf = load_b_bf16(W + (long)(tileN + t * 16 + lidx) * K + k0 + hiB);
      c[t] = __builtin_amdgcn_wmma_f32_16x16x32_bf16(false, a, false, bf,
                                                     (short)0, c[t], false, false);
    }
  }

  const int mOff = (lane >> 4) * 8;  // C/D: VGPR r -> M = mOff + r, N = lidx
#pragma unroll
  for (int t = 0; t < 4; ++t) {
#pragma unroll
    for (int r = 0; r < 8; ++r) {
      const int m = tileM + mOff + r;
      const int n = tileN + t * 16 + lidx;
      const float val = c[t][r] + bias[n];
      if (HEAD_SPLIT) {
        const int b = m >> 11, s = m & (Sn - 1);
        const int h = n >> 6,  d = n & (Dk - 1);
        Yb[((long)(b * Hn + h) * Sn + s) * Dk + d] = (__bf16)val;
      } else {
        Yf[(long)m * N + n] = val;
      }
    }
  }
}

// ---------------------------------------------------------------------------
// Fused attention: one workgroup per (b, h, 16-row q tile).
// Phase 1: scores (16 x 2048) via WMMA -> scale, mask -> LDS (128 KB of the
//          320 KB/WGP CDNA5 LDS).
// Phase 2: wave-level softmax in LDS; stream attn probs (fp32, non-temporal).
// Phase 3: ctx = attn @ V via WMMA, A-fragments from LDS, K split over waves.
// ---------------------------------------------------------------------------
__global__ __launch_bounds__(256) void attn_fused_kernel(
    const __bf16* __restrict__ Qb, const __bf16* __restrict__ Kb,
    const __bf16* __restrict__ Vb, const int* __restrict__ mask,
    float* __restrict__ attn_out, __bf16* __restrict__ ctx) {
  extern __shared__ float sc[];  // 16*Sn scores + 1024 reduction scratch
  const int lane = threadIdx.x & 31;
  const int wave = threadIdx.x >> 5;
  const int q0   = blockIdx.x * 16;
  const int h    = blockIdx.y;
  const int b    = blockIdx.z;
  const long headBase = (long)(b * Hn + h) * Sn * Dk;
  const int lidx = lane & 15;
  const int hiA  = (lane >> 4) * 16;
  const int hiB  = (lane >> 4) * 16;
  const int mOff = (lane >> 4) * 8;

  // ---- Phase 1: scores, 256 key columns per wave (two register passes) ----
  for (int p = 0; p < 2; ++p) {
    const int colBase = wave * 256 + p * 128;
    v8f acc[8] = {v8f{}, v8f{}, v8f{}, v8f{}, v8f{}, v8f{}, v8f{}, v8f{}};
#pragma unroll
    for (int k0 = 0; k0 < Dk; k0 += 32) {
      const v16bf a = load_a_bf16(Qb + headBase + (long)(q0 + lidx) * Dk + k0 + hiA);
#pragma unroll
      for (int t = 0; t < 8; ++t) {
        const int n = colBase + t * 16 + lidx;  // key index = B column
        const v16bf kf = load_b_bf16(Kb + headBase + (long)n * Dk + k0 + hiB);
        acc[t] = __builtin_amdgcn_wmma_f32_16x16x32_bf16(false, a, false, kf,
                                                         (short)0, acc[t], false, false);
      }
    }
#pragma unroll
    for (int t = 0; t < 8; ++t) {
#pragma unroll
      for (int r = 0; r < 8; ++r) {
        const int row = mOff + r;
        const int col = colBase + t * 16 + lidx;
        const float v = acc[t][r] * 0.125f;  // 1/sqrt(64)
        const int  mv = mask[((long)b * Sn + (q0 + row)) * Sn + col];
        sc[row * Sn + col] = (mv == 0) ? -1.0e20f : v;
      }
    }
  }
  __syncthreads();

  // ---- Phase 2: softmax, 2 rows per wave, cross-lane reduce via shuffles ----
  for (int rr = 0; rr < 2; ++rr) {
    const int row = wave * 2 + rr;
    float* rp = sc + row * Sn;
    float mx = -3.0e38f;
    for (int c = lane; c < Sn; c += 32) mx = fmaxf(mx, rp[c]);
#pragma unroll
    for (int off = 16; off >= 1; off >>= 1) mx = fmaxf(mx, __shfl_xor(mx, off, 32));
    float sum = 0.f;
    for (int c = lane; c < Sn; c += 32) {
      const float e = __expf(rp[c] - mx);
      rp[c] = e;
      sum += e;
    }
#pragma unroll
    for (int off = 16; off >= 1; off >>= 1) sum += __shfl_xor(sum, off, 32);
    const float inv = 1.0f / sum;
    const long abase = ((long)(b * Hn + h) * Sn + (q0 + row)) * Sn;
    for (int c = lane; c < Sn; c += 32) {
      const float pv = rp[c] * inv;
      rp[c] = pv;
      __builtin_nontemporal_store(pv, &attn_out[abase + c]);  // streamed output
    }
  }
  __syncthreads();

  // ---- Phase 3: ctx = attn(16x2048) @ V(2048x64); K halves across wave pairs
  const int td    = wave & 3;   // which 16-wide slice of d_k
  const int khalf = wave >> 2;  // which half of the 2048 K range
  v8f cacc = {};
  for (int kk = 0; kk < Sn / 2; kk += 32) {
    const int k0 = khalf * (Sn / 2) + kk;
    v16bf a;
    {
      const float* ap = sc + lidx * Sn + k0 + hiA;  // LDS -> bf16 A fragment
#pragma unroll
      for (int e = 0; e < 16; ++e) a[e] = (__bf16)ap[e];
    }
    v16bf vf;  // B column n = td*16+lidx of V, strided by Dk over K
    {
      const __bf16* vp = Vb + headBase + (long)(k0 + hiB) * Dk + td * 16 + lidx;
#pragma unroll
      for (int e = 0; e < 16; ++e) vf[e] = vp[(long)e * Dk];
    }
    cacc = __builtin_amdgcn_wmma_f32_16x16x32_bf16(false, a, false, vf,
                                                   (short)0, cacc, false, false);
  }
  float* red = sc + 16 * Sn;  // 1024-float scratch beyond the score tile
  if (khalf == 1) {
#pragma unroll
    for (int r = 0; r < 8; ++r) red[td * 256 + r * 32 + lane] = cacc[r];
  }
  __syncthreads();
  if (khalf == 0) {
#pragma unroll
    for (int r = 0; r < 8; ++r) {
      const float val = cacc[r] + red[td * 256 + r * 32 + lane];
      const int s = q0 + mOff + r;
      const int d = h * Dk + td * 16 + lidx;
      ctx[((long)b * Sn + s) * Dm + d] = (__bf16)val;
    }
  }
}

// ---------------------------------------------------------------------------
extern "C" void kernel_launch(void* const* d_in, const int* in_sizes, int n_in,
                              void* d_out, int out_size, void* d_ws, size_t ws_size,
                              hipStream_t stream) {
  const float* q  = (const float*)d_in[0];
  const float* k  = (const float*)d_in[1];
  const float* v  = (const float*)d_in[2];
  const int*   mk = (const int*)d_in[3];
  const float* wq = (const float*)d_in[4];
  const float* bq = (const float*)d_in[5];
  const float* wk = (const float*)d_in[6];
  const float* bk = (const float*)d_in[7];
  const float* wv = (const float*)d_in[8];
  const float* bv = (const float*)d_in[9];
  const float* wo = (const float*)d_in[10];
  const float* bo = (const float*)d_in[11];

  float* out  = (float*)d_out;                 // [B,S,Dm] fp32
  float* attn = out + (long)Bn * Sn * Dm;      // [B,H,S,S] fp32

  const long nX = (long)Bn * Sn * Dm;          // 4,194,304 activation elements
  const long nW = (long)Dm * Dm;               // 1,048,576 weight elements

  __bf16* p    = (__bf16*)d_ws;
  __bf16* Xq   = p; p += nX;                   // bf16 copies of activations
  __bf16* Xk   = p; p += nX;
  __bf16* Xv   = p; p += nX;
  __bf16* Wqb  = p; p += nW;                   // bf16 copies of weights
  __bf16* Wkb  = p; p += nW;
  __bf16* Wvb  = p; p += nW;
  __bf16* Wob  = p; p += nW;
  __bf16* Qb   = p; p += nX;                   // [B,H,S,Dk] head-split Q/K/V
  __bf16* Kb   = p; p += nX;
  __bf16* Vb   = p; p += nX;
  __bf16* Cb   = p; p += nX;                   // ctx [B,S,Dm]

  dim3 blk(256);

  // ---- fp32 -> bf16 pre-conversion (one streamed pass each) ----
  dim3 gcX((unsigned)(nX / 4 / 256));
  dim3 gcW((unsigned)(nW / 4 / 256));
  hipLaunchKernelGGL(f32_to_bf16_kernel, gcX, blk, 0, stream, q,  Xq,  nX);
  hipLaunchKernelGGL(f32_to_bf16_kernel, gcX, blk, 0, stream, k,  Xk,  nX);
  hipLaunchKernelGGL(f32_to_bf16_kernel, gcX, blk, 0, stream, v,  Xv,  nX);
  hipLaunchKernelGGL(f32_to_bf16_kernel, gcW, blk, 0, stream, wq, Wqb, nW);
  hipLaunchKernelGGL(f32_to_bf16_kernel, gcW, blk, 0, stream, wk, Wkb, nW);
  hipLaunchKernelGGL(f32_to_bf16_kernel, gcW, blk, 0, stream, wv, Wvb, nW);
  hipLaunchKernelGGL(f32_to_bf16_kernel, gcW, blk, 0, stream, wo, Wob, nW);

  // ---- Q/K/V projections (WMMA GEMM, head-split bf16 output) ----
  dim3 g1(Dm / 64, (Bn * Sn) / 128);           // 16 x 32 tiles
  hipLaunchKernelGGL((gemm_bf16_kernel<true>), g1, blk, 0, stream,
                     Xq, Wqb, bq, (float*)nullptr, Qb, Bn * Sn, Dm, Dm);
  hipLaunchKernelGGL((gemm_bf16_kernel<true>), g1, blk, 0, stream,
                     Xk, Wkb, bk, (float*)nullptr, Kb, Bn * Sn, Dm, Dm);
  hipLaunchKernelGGL((gemm_bf16_kernel<true>), g1, blk, 0, stream,
                     Xv, Wvb, bv, (float*)nullptr, Vb, Bn * Sn, Dm, Dm);

  // ---- fused attention ----
  dim3 g2(Sn / 16, Hn, Bn);
  const size_t lds = (size_t)(16 * Sn + 1024) * sizeof(float);  // 132 KB < 320 KB
  hipLaunchKernelGGL(attn_fused_kernel, g2, blk, lds, stream,
                     Qb, Kb, Vb, mk, attn, Cb);

  // ---- output projection ----
  hipLaunchKernelGGL((gemm_bf16_kernel<false>), g1, blk, 0, stream,
                     Cb, Wob, bo, out, (__bf16*)nullptr, Bn * Sn, Dm, Dm);
}
